// RNNModel_32332513804947
// MI455X (gfx1250) — compile-verified, matrix operand order checked by reading, not run on hardware
//
#include <hip/hip_runtime.h>

// ---------------------------------------------------------------------------
// CDNA5 (gfx1250) wave32 WMMA implementation of the GRU seq2seq model.
// All GEMMs use v_wmma_f32_16x16x32_f16 (f16 inputs, f32 accumulate).
// Round 3: zero-init accumulators (inline-0 C operand), epilogue biases,
// capped K-loop unrolling to kill residual scratch spills.
// ---------------------------------------------------------------------------

typedef _Float16 v16h __attribute__((ext_vector_type(16)));
typedef float    v8f  __attribute__((ext_vector_type(8)));

#define S_LEN 128
#define BATCH 32
#define T_LEN 64
#define EMB   256
#define HID   512   // encoder hidden H, decoder hidden D (both 512)
#define VTGT  2000
#define NCONST 8

__device__ __forceinline__ v8f wmma16(v16h a, v16h b, v8f c) {
  // D = A(16x32 f16) * B(32x16 f16) + C(16x16 f32)
  return __builtin_amdgcn_wmma_f32_16x16x32_f16(false, a, false, b, (short)0, c,
                                                false, false);
}
__device__ __forceinline__ float sigmoidf_(float x) {
  return 1.0f / (1.0f + __expf(-x));
}

// --------------------------- utility kernels -------------------------------

__global__ void cvt_kernel(const float* __restrict__ src, int srcLd, int colOff,
                           _Float16* __restrict__ dst, int rows, int cols) {
  int i = blockIdx.x * blockDim.x + threadIdx.x;
  if (i >= rows * cols) return;
  int r = i / cols, c = i - r * cols;
  dst[i] = (_Float16)src[r * srcLd + colOff + c];
}

__global__ void embsrc_kernel(const int* __restrict__ questions,
                              const float* __restrict__ src_emb,
                              _Float16* __restrict__ dst) {
  int i = blockIdx.x * blockDim.x + threadIdx.x;
  if (i >= S_LEN * BATCH * EMB) return;
  int e = i & (EMB - 1);
  int sb = i >> 8;                       // s*32 + b
  dst[i] = (_Float16)src_emb[questions[sb] * EMB + e];
}

__global__ void zero_kernel(float* __restrict__ p, int n) {
  int i = blockIdx.x * blockDim.x + threadIdx.x;
  if (i < n) p[i] = 0.0f;
}

// --------------------------- encoder (bidir GRU) ---------------------------
// grid = 2 (dir), block = 1024 threads (32 waves x 2 tile-tasks).
// h lives in ping-pong LDS (2 x 32 KB f16).  Per task only 4 WMMA
// accumulators: r and z chain gi+gh in one accumulator; n keeps gi_n/gh_n.
// Accumulators zero-init (first WMMA uses the inline-0 C operand); biases are
// applied in the epilogue.
__global__ void __launch_bounds__(1024)
encoder_kernel(const _Float16* __restrict__ embsrc,
               const _Float16* __restrict__ WihF,
               const _Float16* __restrict__ WhhF,
               const float* __restrict__ bihF,
               const float* __restrict__ bhhF,
               const _Float16* __restrict__ WihB,
               const _Float16* __restrict__ WhhB,
               const float* __restrict__ bihB,
               const float* __restrict__ bhhB,
               _Float16* __restrict__ encoded /* 128x32x1024 */) {
  __shared__ _Float16 hbuf[2][BATCH * HID];      // 64 KB ping-pong
  const int dir  = blockIdx.x;
  const _Float16* Wih = dir ? WihB : WihF;
  const _Float16* Whh = dir ? WhhB : WhhF;
  const float*    bih = dir ? bihB : bihF;
  const float*    bhh = dir ? bhhB : bhhF;

  const int tid = threadIdx.x, lane = tid & 31, wave = tid >> 5;
  const int idx = lane & 15, ksel = lane >> 4;

  for (int i = tid; i < BATCH * HID; i += blockDim.x) hbuf[0][i] = (_Float16)0.0f;
  __syncthreads();

  int p = 0;
  for (int step = 0; step < S_LEN; ++step) {
    const int s = dir ? (S_LEN - 1 - step) : step;  // actual input index
    const _Float16* x = embsrc + s * BATCH * EMB;
    const _Float16* hcur = hbuf[p];
    _Float16* hnxt = hbuf[p ^ 1];

#pragma clang loop unroll(disable)
    for (int ti = 0; ti < 2; ++ti) {
      const int task = wave * 2 + ti;              // 0..63
      const int rt = task >> 5, nc = task & 31;
      const int col = nc * 16 + idx;
      const _Float16* xp = x + (rt * 16 + idx) * EMB + ksel * 16;
      const _Float16* hp = hcur + (rt * 16 + idx) * HID + ksel * 16;
      const _Float16* wi = Wih + (nc * 16 + idx) * EMB + ksel * 16;
      const _Float16* wh = Whh + (nc * 16 + idx) * HID + ksel * 16;

      v8f accr = {}, accz = {}, gin = {}, ghn = {};
#pragma clang loop unroll_count(2)
      for (int kk = 0; kk < EMB; kk += 32) {
        v16h a = *(const v16h*)(xp + kk);
        accr = wmma16(a, *(const v16h*)(wi + kk), accr);
        accz = wmma16(a, *(const v16h*)(wi + kk + 512 * EMB), accz);
        gin  = wmma16(a, *(const v16h*)(wi + kk + 1024 * EMB), gin);
      }
#pragma clang loop unroll_count(2)
      for (int kk = 0; kk < HID; kk += 32) {
        v16h a = *(const v16h*)(hp + kk);
        accr = wmma16(a, *(const v16h*)(wh + kk), accr);
        accz = wmma16(a, *(const v16h*)(wh + kk + 512 * HID), accz);
        ghn  = wmma16(a, *(const v16h*)(wh + kk + 1024 * HID), ghn);
      }
      const float b_r  = bih[col] + bhh[col];
      const float b_z  = bih[512 + col] + bhh[512 + col];
      const float b_in = bih[1024 + col];
      const float b_hn = bhh[1024 + col];
      for (int v = 0; v < 8; ++v) {
        const int row = rt * 16 + ksel * 8 + v;
        float r = sigmoidf_(accr[v] + b_r);
        float z = sigmoidf_(accz[v] + b_z);
        float n = tanhf(gin[v] + b_in + r * (ghn[v] + b_hn));
        float hold = (float)hcur[row * HID + col];
        _Float16 hv = (_Float16)((1.0f - z) * n + z * hold);
        hnxt[row * HID + col] = hv;
        encoded[(s * BATCH + row) * (2 * HID) + dir * HID + col] = hv;
      }
    }
    __syncthreads();
    p ^= 1;
  }
}

// --------------------------- enc_proj --------------------------------------
// enc_proj = tanh([hf_last, hb_last] @ fc_W^T + fc_b).  hf_last = encoded[127,:,:512],
// hb_last = encoded[0,:,512:].  One block, 16 waves x 4 tiles, K = 1024.
__global__ void enc_proj_kernel(const _Float16* __restrict__ encoded,
                                const _Float16* __restrict__ fcW16,
                                const float* __restrict__ fcB,
                                float* __restrict__ proj32, _Float16* __restrict__ proj16,
                                float* __restrict__ hdec32, _Float16* __restrict__ hdec16) {
  const int tid = threadIdx.x, lane = tid & 31, wave = tid >> 5;
  const int idx = lane & 15, ksel = lane >> 4;
#pragma clang loop unroll(disable)
  for (int task = wave * 4; task < wave * 4 + 4; ++task) {
    const int rt = task >> 5, nc = task & 31;
    const int b = rt * 16 + idx;
    const _Float16* pf = encoded + ((S_LEN - 1) * BATCH + b) * (2 * HID);
    const _Float16* pb = encoded + (b) * (2 * HID);
    const _Float16* wp = fcW16 + (nc * 16 + idx) * (2 * HID) + ksel * 16;
    v8f acc = {};
    for (int kk = 0; kk < 2 * HID; kk += 32) {
      const int seg = kk + ksel * 16;
      const _Float16* xp = (seg < HID) ? pf : pb;
      acc = wmma16(*(const v16h*)(xp + seg), *(const v16h*)(wp + kk), acc);
    }
    const int col = nc * 16 + idx;
    for (int v = 0; v < 8; ++v) {
      const int row = rt * 16 + ksel * 8 + v;
      float t = tanhf(acc[v] + fcB[col]);
      proj32[row * HID + col] = t;
      hdec32[row * HID + col] = t;            // initial decoder carry h
      _Float16 th = (_Float16)t;
      proj16[row * HID + col] = th;
      hdec16[row * HID + col] = th;
    }
  }
}

// --------------------------- enc_att (big GEMM) ----------------------------
// enc_att[b,s,:] = enc_p[b,s,:] @ Wa_e^T + attn_b  (bias pre-folded).
// 8192 tiles, 8 waves/block -> 1024 blocks.
__global__ void enc_att_kernel(const _Float16* __restrict__ encoded,
                               const _Float16* __restrict__ WaE16,
                               const float* __restrict__ attnB,
                               float* __restrict__ enc_att /* (b*128+s)*512 */) {
  const int tile = blockIdx.x * 8 + (threadIdx.x >> 5);
  const int lane = threadIdx.x & 31, idx = lane & 15, ksel = lane >> 4;
  const int mt = tile >> 5, nt = tile & 31;
  const int r = mt * 16 + idx;                 // row = b*128 + s
  const int b = r >> 7, s = r & 127;
  const _Float16* xp = encoded + (s * BATCH + b) * (2 * HID) + ksel * 16;
  const _Float16* wp = WaE16 + (nt * 16 + idx) * (2 * HID) + ksel * 16;
  v8f acc = {};
  for (int kk = 0; kk < 2 * HID; kk += 32)
    acc = wmma16(*(const v16h*)(xp + kk), *(const v16h*)(wp + kk), acc);
  const int col = nt * 16 + idx;
  for (int v = 0; v < 8; ++v) {
    const int rr = mt * 16 + ksel * 8 + v;
    enc_att[rr * HID + col] = acc[v] + attnB[col];
  }
}

// --------------------------- gh_dec (constant!) ----------------------------
// Decoder GRU uses enc_proj as hidden every step -> gh = enc_proj@dec_Whh^T + bhh
// is step-invariant.  192 tiles (2 x 96), grid 24 x 256.
__global__ void ghdec_kernel(const _Float16* __restrict__ proj16,
                             const _Float16* __restrict__ decWhh16,
                             const float* __restrict__ dBhh,
                             float* __restrict__ ghdec /* 32x1536 */) {
  const int tile = blockIdx.x * 8 + (threadIdx.x >> 5);
  const int lane = threadIdx.x & 31, idx = lane & 15, ksel = lane >> 4;
  const int rt = tile / 96, nc = tile % 96;
  const _Float16* xp = proj16 + (rt * 16 + idx) * HID + ksel * 16;
  const _Float16* wp = decWhh16 + (nc * 16 + idx) * HID + ksel * 16;
  v8f acc = {};
  for (int kk = 0; kk < HID; kk += 32)
    acc = wmma16(*(const v16h*)(xp + kk), *(const v16h*)(wp + kk), acc);
  const int col = nc * 16 + idx;
  for (int v = 0; v < 8; ++v) {
    const int row = rt * 16 + ksel * 8 + v;
    ghdec[row * 1536 + col] = acc[v] + dBhh[col];
  }
}

// --------------------------- attention step --------------------------------
// One block, 512 threads. WMMA h@Wa_h^T (hW in LDS f16), then wave-shuffle
// energy/softmax, weighted sum, copy weights, p_gen, emb gather.
__global__ void attn_kernel(const _Float16* __restrict__ hdec16,
                            const float* __restrict__ hdec32,
                            const _Float16* __restrict__ WaH16,
                            const float* __restrict__ enc_att,
                            const float* __restrict__ vW,
                            const _Float16* __restrict__ encoded,
                            const int* __restrict__ align_,
                            const int* __restrict__ prev_tok,
                            const float* __restrict__ tgt_emb,
                            const float* __restrict__ proj32,
                            const float* __restrict__ copyW,
                            const float* __restrict__ copyB,
                            _Float16* __restrict__ xcat   /* 32x1280 */,
                            _Float16* __restrict__ outcat /* 32x1792 */,
                            float* __restrict__ pgen,
                            float* __restrict__ copyw /* 32x8 */) {
  __shared__ _Float16 hW[BATCH * HID];     // 32 KB
  __shared__ float    sc[BATCH * S_LEN];   // 16 KB
  __shared__ float    cw[BATCH * NCONST];

  const int tid = threadIdx.x, lane = tid & 31, wave = tid >> 5;
  const int idx = lane & 15, ksel = lane >> 4;

  // target-embedding gather -> xcat[:,0:256], outcat[:,1536:1792]
  for (int i = tid; i < BATCH * EMB; i += blockDim.x) {
    int b = i >> 8, e = i & (EMB - 1);
    _Float16 hv = (_Float16)tgt_emb[prev_tok[b] * EMB + e];
    xcat[b * 1280 + e] = hv;
    outcat[b * 1792 + 1536 + e] = hv;
  }

  // hW = h @ Wa_h^T (32x512), 64 tiles
#pragma clang loop unroll(disable)
  for (int task = wave * 4; task < wave * 4 + 4; ++task) {
    const int rt = task >> 5, nc = task & 31;
    const _Float16* xp = hdec16 + (rt * 16 + idx) * HID + ksel * 16;
    const _Float16* wp = WaH16 + (nc * 16 + idx) * HID + ksel * 16;
    v8f acc = {};
    for (int kk = 0; kk < HID; kk += 32)
      acc = wmma16(*(const v16h*)(xp + kk), *(const v16h*)(wp + kk), acc);
    const int col = nc * 16 + idx;
    for (int v = 0; v < 8; ++v)
      hW[(rt * 16 + ksel * 8 + v) * HID + col] = (_Float16)acc[v];
  }
  __syncthreads();

  // energies: one wave per (b,s) pair, shuffle-reduced dot with v_w
  for (int p = wave; p < BATCH * S_LEN; p += 16) {
    const int b = p >> 7, s = p & 127;
    const float* ea = enc_att + (b * S_LEN + s) * HID;
    const _Float16* hw = hW + b * HID;
    float sum = 0.0f;
    for (int d = lane; d < HID; d += 32)
      sum += vW[d] * tanhf(ea[d] + (float)hw[d]);
    for (int m = 16; m; m >>= 1) sum += __shfl_xor(sum, m, 32);
    if (lane == 0) sc[b * S_LEN + s] = sum;
  }
  __syncthreads();

  // softmax over s per batch row
  for (int b = wave; b < BATCH; b += 16) {
    float mx = -1e30f;
    for (int s = lane; s < S_LEN; s += 32) mx = fmaxf(mx, sc[b * S_LEN + s]);
    for (int m = 16; m; m >>= 1) mx = fmaxf(mx, __shfl_xor(mx, m, 32));
    float sum = 0.0f;
    for (int s = lane; s < S_LEN; s += 32) {
      float e = __expf(sc[b * S_LEN + s] - mx);
      sc[b * S_LEN + s] = e;
      sum += e;
    }
    for (int m = 16; m; m >>= 1) sum += __shfl_xor(sum, m, 32);
    float inv = 1.0f / sum;
    for (int s = lane; s < S_LEN; s += 32) sc[b * S_LEN + s] *= inv;
  }
  __syncthreads();

  // weighted[b,e] = sum_s sc[b,s] * encoded[s,b,e]   (e < 1024)
  for (int i = tid; i < BATCH * 2 * HID; i += blockDim.x) {
    int b = i >> 10, e = i & 1023;
    float acc = 0.0f;
    for (int s = 0; s < S_LEN; ++s)
      acc += sc[b * S_LEN + s] * (float)encoded[(s * BATCH + b) * (2 * HID) + e];
    _Float16 hv = (_Float16)acc;
    xcat[b * 1280 + EMB + e] = hv;
    outcat[b * 1792 + HID + e] = hv;
  }

  // tok_att / copy weights
  for (int i = tid; i < BATCH * NCONST; i += blockDim.x) {
    int b = i >> 3, c = i & 7;
    const int* al = align_ + (b * NCONST + c) * S_LEN;
    float acc = 0.0f;
    for (int s = 0; s < S_LEN; ++s) acc += sc[b * S_LEN + s] * (float)al[s];
    cw[i] = acc;
  }
  __syncthreads();
  for (int b = tid; b < BATCH; b += blockDim.x) {
    float s = 0.0f;
    for (int c = 0; c < NCONST; ++c) s += cw[b * NCONST + c];
    float inv = 1.0f / s;
    for (int c = 0; c < NCONST; ++c) copyw[b * NCONST + c] = cw[b * NCONST + c] * inv;
  }

  // p_gen = sigmoid([enc_proj, h, emb] . copy_W + copy_b), one wave per b
  for (int b = wave; b < BATCH; b += 16) {
    float sum = 0.0f;
    for (int k = lane; k < 1280; k += 32) {
      float xv = (k < 512) ? proj32[b * HID + k]
               : (k < 1024) ? hdec32[b * HID + (k - 512)]
                            : tgt_emb[prev_tok[b] * EMB + (k - 1024)];
      sum += copyW[k] * xv;
    }
    for (int m = 16; m; m >>= 1) sum += __shfl_xor(sum, m, 32);
    if (lane == 0) pgen[b] = sigmoidf_(sum + copyB[0]);
  }
}

// --------------------------- decoder GRU step ------------------------------
// gi = xcat @ dec_Wih^T (K=1280) only; gh is the precomputed ghdec.
// Zero-init accumulators; bih/ghdec applied in the epilogue.
__global__ void dec_gru_kernel(const _Float16* __restrict__ xcat,
                               const _Float16* __restrict__ decWih16,
                               const float* __restrict__ dBih,
                               const float* __restrict__ ghdec,
                               const float* __restrict__ proj32,
                               float* __restrict__ hdec32,
                               _Float16* __restrict__ hdec16,
                               _Float16* __restrict__ outcat) {
  const int tid = threadIdx.x, lane = tid & 31, wave = tid >> 5;
  const int idx = lane & 15, ksel = lane >> 4;
#pragma clang loop unroll(disable)
  for (int task = wave * 4; task < wave * 4 + 4; ++task) {
    const int rt = task >> 5, nc = task & 31;
    const int col = nc * 16 + idx;
    const int row0 = rt * 16 + ksel * 8;
    const _Float16* xp = xcat + (rt * 16 + idx) * 1280 + ksel * 16;
    const _Float16* wp = decWih16 + (nc * 16 + idx) * 1280 + ksel * 16;

    v8f gr = {}, gz = {}, gn = {};
#pragma clang loop unroll_count(2)
    for (int kk = 0; kk < 1280; kk += 32) {
      v16h a = *(const v16h*)(xp + kk);
      gr = wmma16(a, *(const v16h*)(wp + kk), gr);
      gz = wmma16(a, *(const v16h*)(wp + kk + 512 * 1280), gz);
      gn = wmma16(a, *(const v16h*)(wp + kk + 1024 * 1280), gn);
    }
    const float b_r = dBih[col], b_z = dBih[512 + col], b_n = dBih[1024 + col];
    for (int v = 0; v < 8; ++v) {
      const int row = row0 + v;
      float r = sigmoidf_(gr[v] + b_r + ghdec[row * 1536 + col]);
      float z = sigmoidf_(gz[v] + b_z + ghdec[row * 1536 + 512 + col]);
      float n = tanhf(gn[v] + b_n + r * ghdec[row * 1536 + 1024 + col]);
      float hn = (1.0f - z) * n + z * proj32[row * HID + col];
      hdec32[row * HID + col] = hn;
      _Float16 hh = (_Float16)hn;
      hdec16[row * HID + col] = hh;
      outcat[row * 1792 + col] = hh;
    }
  }
}

// --------------------------- output GEMM -----------------------------------
// logits(32x2000) = outcat(32x1792) @ out_W^T + out_b.  250 tiles, K=56 WMMAs.
__global__ void out_gemm_kernel(const _Float16* __restrict__ outcat,
                                const _Float16* __restrict__ outW16,
                                const float* __restrict__ outB,
                                float* __restrict__ logits) {
  const int tile = blockIdx.x * 8 + (threadIdx.x >> 5);
  if (tile >= 250) return;                 // wave-uniform guard, EXEC stays full
  const int lane = threadIdx.x & 31, idx = lane & 15, ksel = lane >> 4;
  const int mt = tile / 125, nt = tile % 125;
  const _Float16* xp = outcat + (mt * 16 + idx) * 1792 + ksel * 16;
  const _Float16* wp = outW16 + (nt * 16 + idx) * 1792 + ksel * 16;
  v8f acc = {};
  for (int kk = 0; kk < 1792; kk += 32)
    acc = wmma16(*(const v16h*)(xp + kk), *(const v16h*)(wp + kk), acc);
  const int col = nt * 16 + idx;
  for (int v = 0; v < 8; ++v) {
    const int row = mt * 16 + ksel * 8 + v;
    logits[row * VTGT + col] = acc[v] + outB[col];
  }
}

// --------------------------- finalize (softmax + copy mix) -----------------
__global__ void finalize_kernel(const float* __restrict__ logits,
                                const float* __restrict__ pgen,
                                const float* __restrict__ copyw,
                                const int* __restrict__ const_map,
                                float* __restrict__ outrow /* 32x2000 */) {
  const int gwave = (blockIdx.x * blockDim.x + threadIdx.x) >> 5;
  const int lane = threadIdx.x & 31;
  if (gwave >= BATCH) return;
  const int b = gwave;
  const float* l = logits + b * VTGT;
  int cm[NCONST];
  for (int j = 0; j < NCONST; ++j) cm[j] = const_map[j];

  float mx = -1e30f;
  for (int v = lane; v < VTGT; v += 32) mx = fmaxf(mx, l[v]);
  for (int m = 16; m; m >>= 1) mx = fmaxf(mx, __shfl_xor(mx, m, 32));
  float sum = 0.0f;
  for (int v = lane; v < VTGT; v += 32) sum += __expf(l[v] - mx);
  for (int m = 16; m; m >>= 1) sum += __shfl_xor(sum, m, 32);
  const float inv = 1.0f / sum;
  const float pg = pgen[b];
  for (int v = lane; v < VTGT; v += 32) {
    float cp = 0.0f;
    for (int j = 0; j < NCONST; ++j)
      if (cm[j] == v) cp += copyw[b * NCONST + j];
    outrow[b * VTGT + v] = pg * __expf(l[v] - mx) * inv + (1.0f - pg) * cp;
  }
}

// ---------------------------------------------------------------------------

extern "C" void kernel_launch(void* const* d_in, const int* in_sizes, int n_in,
                              void* d_out, int out_size, void* d_ws, size_t ws_size,
                              hipStream_t stream) {
  (void)in_sizes; (void)n_in; (void)out_size; (void)ws_size;
  const int*   questions = (const int*)d_in[0];
  const int*   equations = (const int*)d_in[1];
  const int*   align_    = (const int*)d_in[2];
  const int*   const_map = (const int*)d_in[3];
  const float* src_emb   = (const float*)d_in[4];
  const float* tgt_emb   = (const float*)d_in[5];
  const float* eWihF = (const float*)d_in[6];
  const float* eWhhF = (const float*)d_in[7];
  const float* eBihF = (const float*)d_in[8];
  const float* eBhhF = (const float*)d_in[9];
  const float* eWihB = (const float*)d_in[10];
  const float* eWhhB = (const float*)d_in[11];
  const float* eBihB = (const float*)d_in[12];
  const float* eBhhB = (const float*)d_in[13];
  const float* fcW   = (const float*)d_in[14];
  const float* fcB   = (const float*)d_in[15];
  const float* attnW = (const float*)d_in[16];
  const float* attnB = (const float*)d_in[17];
  const float* vW    = (const float*)d_in[18];
  const float* dWih  = (const float*)d_in[19];
  const float* dWhh  = (const float*)d_in[20];
  const float* dBih  = (const float*)d_in[21];
  const float* dBhh  = (const float*)d_in[22];
  const float* outW  = (const float*)d_in[23];
  const float* outB  = (const float*)d_in[24];
  const float* copyW = (const float*)d_in[25];
  const float* copyB = (const float*)d_in[26];
  float* out = (float*)d_out;

  // workspace bump allocator (256B aligned)
  char* base = (char*)d_ws;
  size_t off = 0;
  auto alloc = [&](size_t bytes) -> void* {
    void* r = base + off;
    off = (off + bytes + 255) & ~(size_t)255;
    return r;
  };
  _Float16* wihF16   = (_Float16*)alloc((size_t)1536 * 256 * 2);
  _Float16* whhF16   = (_Float16*)alloc((size_t)1536 * 512 * 2);
  _Float16* wihB16   = (_Float16*)alloc((size_t)1536 * 256 * 2);
  _Float16* whhB16   = (_Float16*)alloc((size_t)1536 * 512 * 2);
  _Float16* fcW16    = (_Float16*)alloc((size_t)512 * 1024 * 2);
  _Float16* WaH16    = (_Float16*)alloc((size_t)512 * 512 * 2);
  _Float16* WaE16    = (_Float16*)alloc((size_t)512 * 1024 * 2);
  _Float16* decWih16 = (_Float16*)alloc((size_t)1536 * 1280 * 2);
  _Float16* decWhh16 = (_Float16*)alloc((size_t)1536 * 512 * 2);
  _Float16* outW16   = (_Float16*)alloc((size_t)2000 * 1792 * 2);
  _Float16* embsrc16 = (_Float16*)alloc((size_t)S_LEN * BATCH * EMB * 2);
  _Float16* encoded  = (_Float16*)alloc((size_t)S_LEN * BATCH * 1024 * 2);
  float*    proj32   = (float*)alloc((size_t)BATCH * HID * 4);
  _Float16* proj16   = (_Float16*)alloc((size_t)BATCH * HID * 2);
  float*    hdec32   = (float*)alloc((size_t)BATCH * HID * 4);
  _Float16* hdec16   = (_Float16*)alloc((size_t)BATCH * HID * 2);
  float*    ghdec    = (float*)alloc((size_t)BATCH * 1536 * 4);
  _Float16* xcat     = (_Float16*)alloc((size_t)BATCH * 1280 * 2);
  _Float16* outcat   = (_Float16*)alloc((size_t)BATCH * 1792 * 2);
  float*    pgen     = (float*)alloc((size_t)BATCH * 4);
  float*    copyw    = (float*)alloc((size_t)BATCH * NCONST * 4);
  float*    logits   = (float*)alloc((size_t)BATCH * VTGT * 4);
  float*    enc_att  = (float*)alloc((size_t)BATCH * S_LEN * HID * 4);

  auto cvt = [&](const float* src, int ld, int coff, _Float16* dst, int rows, int cols) {
    int n = rows * cols;
    cvt_kernel<<<(n + 255) / 256, 256, 0, stream>>>(src, ld, coff, dst, rows, cols);
  };
  // one-time f32 -> f16 weight conversion (L2-resident after first touch)
  cvt(eWihF, 256, 0, wihF16, 1536, 256);
  cvt(eWhhF, 512, 0, whhF16, 1536, 512);
  cvt(eWihB, 256, 0, wihB16, 1536, 256);
  cvt(eWhhB, 512, 0, whhB16, 1536, 512);
  cvt(fcW, 1024, 0, fcW16, 512, 1024);
  cvt(attnW, 1536, 0, WaH16, 512, 512);     // attn_W[:, :512]
  cvt(attnW, 1536, 512, WaE16, 512, 1024);  // attn_W[:, 512:]
  cvt(dWih, 1280, 0, decWih16, 1536, 1280);
  cvt(dWhh, 512, 0, decWhh16, 1536, 512);
  cvt(outW, 1792, 0, outW16, 2000, 1792);

  {
    int n = S_LEN * BATCH * EMB;
    embsrc_kernel<<<(n + 255) / 256, 256, 0, stream>>>(questions, src_emb, embsrc16);
  }

  // bidirectional encoder GRU (2 WGs, sequential 128 steps each)
  encoder_kernel<<<2, 1024, 0, stream>>>(embsrc16, wihF16, whhF16, eBihF, eBhhF,
                                         wihB16, whhB16, eBihB, eBhhB, encoded);

  enc_proj_kernel<<<1, 512, 0, stream>>>(encoded, fcW16, fcB, proj32, proj16,
                                         hdec32, hdec16);
  enc_att_kernel<<<1024, 256, 0, stream>>>(encoded, WaE16, attnB, enc_att);
  ghdec_kernel<<<24, 256, 0, stream>>>(proj16, decWhh16, dBhh, ghdec);

  // output row 0 = zeros
  {
    int n = BATCH * VTGT;
    zero_kernel<<<(n + 255) / 256, 256, 0, stream>>>(out, n);
  }

  // decoder: 63 sequential steps, 4 stream-ordered kernels each
  for (int t = 1; t < T_LEN; ++t) {
    const int* prev_tok = equations + (t - 1) * BATCH;
    attn_kernel<<<1, 512, 0, stream>>>(hdec16, hdec32, WaH16, enc_att, vW,
                                       encoded, align_, prev_tok, tgt_emb,
                                       proj32, copyW, copyB, xcat, outcat,
                                       pgen, copyw);
    dec_gru_kernel<<<1, 512, 0, stream>>>(xcat, decWih16, dBih, ghdec, proj32,
                                          hdec32, hdec16, outcat);
    out_gemm_kernel<<<32, 256, 0, stream>>>(outcat, outW16, outB, logits);
    finalize_kernel<<<4, 256, 0, stream>>>(logits, pgen, copyw, const_map,
                                           out + (size_t)t * BATCH * VTGT);
  }
}
